// MultiHeadAttention_88923002896848
// MI455X (gfx1250) — matrix operand order repaired and checked by reading, not compiled, not run on hardware
//
#include <hip/hip_runtime.h>
#include <hip/hip_bf16.h>
#include <stdint.h>

// Problem constants (from reference)
#define B_   128
#define T1_  188
#define LC_  70
#define T2_  258      // LC + T1
#define F_   512
#define H_   8
#define DK_  64
#define T2P_ 288      // T2 padded to multiple of 32 (WMMA K-step)
#define INFV 10000.0f

typedef __bf16 bf16_t;
typedef __attribute__((ext_vector_type(16))) __bf16 v16bf;
typedef __attribute__((ext_vector_type(8)))  float  v8f;

union B16Frag { uint32_t u[8]; v16bf v; };

// ---------------------------------------------------------------------------
// Fragment loaders per CDNA5 ISA VGPR layouts (CDNA5 ISA §7.12.2)
// A: 16x32 bf16, M = lane%16 (both halves), K striped over 8 VGPRs
// B: 32x16 bf16, N = lane%16, K striped over 8 VGPRs same as A
// C/D f32: VGPR r -> M = r + 8*(lane/16), N = lane%16
// Per-lane data is two contiguous 16B chunks -> compiler fuses to b128 loads.
// ---------------------------------------------------------------------------
__device__ __forceinline__ v16bf load_frag_A(const bf16_t* __restrict__ base,
                                             int row0, int maxrow, int stride,
                                             int kbase, int lane) {
  int m = row0 + (lane & 15);
  if (m > maxrow) m = maxrow;
  int h = lane >> 4;
  B16Frag f;
#pragma unroll
  for (int v = 0; v < 8; ++v) {
    int kl = ((v & 4) ? 16 : 0) + h * 8 + (v & 3) * 2;
    f.u[v] = *(const uint32_t*)(base + (size_t)m * stride + kbase + kl);
  }
  return f.v;
}

__device__ __forceinline__ v16bf load_frag_B(const bf16_t* __restrict__ base,
                                             int row0, int maxrow, int stride,
                                             int kbase, int lane) {
  int n = row0 + (lane & 15);
  if (n > maxrow) n = maxrow;
  int h = lane >> 4;
  B16Frag f;
#pragma unroll
  for (int v = 0; v < 8; ++v) {
    int kl = ((v & 4) ? 16 : 0) + h * 8 + (v & 3) * 2;
    f.u[v] = *(const uint32_t*)(base + (size_t)n * stride + kbase + kl);
  }
  return f.v;
}

__device__ __forceinline__ v8f wmma_bf16(v16bf a, v16bf b, v8f c) {
  return __builtin_amdgcn_wmma_f32_16x16x32_bf16(false, a, false, b,
                                                 (short)0, c, false, false);
}

// ---------------------------------------------------------------------------
// fp32 -> bf16 convert
// ---------------------------------------------------------------------------
__global__ void cvt_f32_bf16_kernel(const float* __restrict__ x,
                                    bf16_t* __restrict__ y, int n) {
  int i = blockIdx.x * blockDim.x + threadIdx.x;
  if (i < n) y[i] = (bf16_t)x[i];
}

// Build concat(cache, key_in) and concat(cache, value_in) in bf16, (B,T2,F)
__global__ void pack_kv_kernel(const float* __restrict__ cache,
                               const float* __restrict__ key_in,
                               const float* __restrict__ value_in,
                               bf16_t* __restrict__ xk,
                               bf16_t* __restrict__ xv) {
  int i = blockIdx.x * blockDim.x + threadIdx.x;
  const int total = B_ * T2_ * F_;
  if (i >= total) return;
  int f  = i % F_;
  int bt = i / F_;
  int t  = bt % T2_;
  int b  = bt / T2_;
  float kvv, vvv;
  if (t < LC_) {
    size_t src = ((size_t)b * LC_ + t) * F_ + f;
    kvv = cache[src];
    vvv = cache[src];
  } else {
    size_t src = ((size_t)b * T1_ + (t - LC_)) * F_ + f;
    kvv = key_in[src];
    vvv = value_in[src];
  }
  xk[i] = (bf16_t)kvv;
  xv[i] = (bf16_t)vvv;
}

// ---------------------------------------------------------------------------
// WMMA GEMM: Y[m,n] = sum_k X[m,k] * W[n,k] + bias[n]
// One wave computes a 16x64 strip (4 accumulators) so each A fragment is
// reused across 4 WMMAs. K = F_ = 512 -> 16 k-steps.
// MODE 0: Q  -> bf16 (B,H,T1,DK)
// MODE 1: K  -> bf16 (B,H,T2,DK)
// MODE 2: V  -> bf16 (B,H,DK,T2P)   (transposed, padded rows)
// MODE 3: O  -> f32  (B*T1, F) plain row-major
// ---------------------------------------------------------------------------
template <int MODE>
__global__ void __launch_bounds__(256)
gemm_wmma_kernel(const bf16_t* __restrict__ X, const bf16_t* __restrict__ W,
                 const float* __restrict__ bias, void* __restrict__ Y) {
  constexpr int T = (MODE == 0 || MODE == 3) ? T1_ : T2_;
  constexpr int Mrows = B_ * T;
  constexpr int NStrips = F_ / 64;                       // 8
  constexpr int TotalStrips = (Mrows / 16) * NStrips;

  int lane = threadIdx.x & 31;
  int strip = blockIdx.x * (blockDim.x >> 5) + (threadIdx.x >> 5); // uniform
  if (strip >= TotalStrips) return;
  int mt = strip / NStrips;
  int ng = strip - mt * NStrips;                         // 64-col group

  v8f acc[4] = {};
#pragma unroll 2
  for (int kb = 0; kb < F_; kb += 32) {
    v16bf a = load_frag_A(X, mt * 16, Mrows - 1, F_, kb, lane);
#pragma unroll
    for (int i = 0; i < 4; ++i) {
      v16bf bf = load_frag_B(W, ng * 64 + i * 16, F_ - 1, F_, kb, lane);
      acc[i] = wmma_bf16(a, bf, acc[i]);
    }
  }

  int h2 = lane >> 4;
  int nl = lane & 15;
#pragma unroll
  for (int i = 0; i < 4; ++i) {
    int col = ng * 64 + i * 16 + nl;
    float bval = bias[col];
#pragma unroll
    for (int r = 0; r < 8; ++r) {
      int row = mt * 16 + r + 8 * h2;
      float y = acc[i][r] + bval;
      if (MODE == 3) {
        ((float*)Y)[(size_t)row * F_ + col] = y;
      } else {
        int b_ = row / T;            // constexpr divisor -> mul/shift
        int t_ = row - b_ * T;
        int hh = col >> 6;           // DK = 64
        int dk = col & 63;
        size_t idx;
        if (MODE == 2) idx = (((size_t)b_ * H_ + hh) * DK_ + dk) * T2P_ + t_;
        else           idx = (((size_t)b_ * H_ + hh) * T + t_) * DK_ + dk;
        ((bf16_t*)Y)[idx] = (bf16_t)y;
      }
    }
  }
}

// ---------------------------------------------------------------------------
// Attention: one wave (32 threads) per (b, h, 16-query-row tile)
// S = (Q K^T)/8 -> LDS (f32), masked (branch-free selects); softmax per row
// split over lane pairs (shfl_xor 16); P (bf16) -> LDS;
// ctx = P @ V with LDS A-fragment reused across 4 DK tiles.
// ---------------------------------------------------------------------------
__global__ void __launch_bounds__(32)
attn_kernel(const bf16_t* __restrict__ Q, const bf16_t* __restrict__ Kt,
            const bf16_t* __restrict__ VT, const uint8_t* __restrict__ mask,
            bf16_t* __restrict__ ctx) {
  __shared__ float  S[16][T2P_];   // 18432 B
  __shared__ bf16_t P[16][T2P_];   //  9216 B

  int lane = threadIdx.x;
  int bh = blockIdx.x;             // b*H + h
  int qt = blockIdx.y;             // query tile
  int b  = bh / H_;
  int h  = bh - b * H_;
  int q0 = qt * 16;

  const bf16_t* Qbase = Q  + (size_t)bh * T1_ * DK_;
  const bf16_t* Kbase = Kt + (size_t)bh * T2_ * DK_;
  const bf16_t* Vbase = VT + (size_t)bh * DK_ * T2P_;
  const uint8_t* Mbase = mask + (size_t)b * T1_ * T2_;

  int h2 = lane >> 4;
  int nl = lane & 15;

  // ---- scores: Q fragments hoisted, 18 key tiles x 2 k-steps ----
  v16bf qa0 = load_frag_A(Qbase, q0, T1_ - 1, DK_, 0,  lane);
  v16bf qa1 = load_frag_A(Qbase, q0, T1_ - 1, DK_, 32, lane);

  for (int j = 0; j < T2P_ / 16; ++j) {
    v16bf kb0 = load_frag_B(Kbase, j * 16, T2_ - 1, DK_, 0,  lane);
    v16bf kb1 = load_frag_B(Kbase, j * 16, T2_ - 1, DK_, 32, lane);
    v8f acc = {};
    acc = wmma_bf16(qa0, kb0, acc);
    acc = wmma_bf16(qa1, kb1, acc);

    int t2  = j * 16 + nl;
    int t2c = (t2 < T2_) ? t2 : (T2_ - 1);
#pragma unroll
    for (int r = 0; r < 8; ++r) {
      int m = r + 8 * h2;
      int qrow = q0 + m;
      int qc = (qrow < T1_) ? qrow : (T1_ - 1);
      uint8_t mk = Mbase[(size_t)qc * T2_ + t2c];        // unconditional load
      float s = acc[r] * 0.125f;                         // 1/sqrt(64)
      bool valid = (qrow < T1_) & (t2 < T2_) & (mk != 0);
      S[m][t2] = valid ? s : -INFV;                      // v_cndmask, no branch
    }
  }
  __syncthreads();

  // ---- softmax: 2 lanes per row, halves combined with shfl_xor(16) ----
  {
    int row = nl;
    int t0  = h2 * (T2P_ / 2);
    float mx = -INFV;
    for (int t = 0; t < T2P_ / 2; ++t) mx = fmaxf(mx, S[row][t0 + t]);
    mx = fmaxf(mx, __shfl_xor(mx, 16, 32));
    float sum = 0.0f;
    for (int t = 0; t < T2P_ / 2; ++t) {
      float s = S[row][t0 + t];
      float e = (s <= -0.5f * INFV) ? 0.0f : __expf(s - mx);
      S[row][t0 + t] = e;
      sum += e;
    }
    sum += __shfl_xor(sum, 16, 32);
    float inv = 1.0f / fmaxf(sum, 1e-20f);
    for (int t = 0; t < T2P_ / 2; ++t)
      P[row][t0 + t] = (bf16_t)(S[row][t0 + t] * inv);
  }
  __syncthreads();

  // ---- ctx = P (16 x T2P) @ V (T2P x DK): A frag reused across 4 tiles ----
  const bf16_t* Pl = &P[0][0];
  v8f acc[4] = {};
  for (int j2 = 0; j2 < T2P_ / 32; ++j2) {
    v16bf a = load_frag_A(Pl, 0, 15, T2P_, j2 * 32, lane);
#pragma unroll
    for (int ntile = 0; ntile < 4; ++ntile) {
      v16bf bf = load_frag_B(Vbase, ntile * 16, DK_ - 1, T2P_, j2 * 32, lane);
      acc[ntile] = wmma_bf16(a, bf, acc[ntile]);
    }
  }
#pragma unroll
  for (int ntile = 0; ntile < 4; ++ntile) {
#pragma unroll
    for (int r = 0; r < 8; ++r) {
      int m = r + 8 * h2;
      int qrow = q0 + m;
      if (qrow < T1_)
        ctx[((size_t)b * T1_ + qrow) * F_ + h * DK_ + ntile * 16 + nl] =
            (bf16_t)acc[ntile][r];
    }
  }
}

// ---------------------------------------------------------------------------
extern "C" void kernel_launch(void* const* d_in, const int* in_sizes, int n_in,
                              void* d_out, int out_size, void* d_ws, size_t ws_size,
                              hipStream_t stream) {
  (void)in_sizes; (void)n_in; (void)out_size; (void)ws_size;

  const float*   query    = (const float*)d_in[0];
  const float*   key_in   = (const float*)d_in[1];
  const float*   value_in = (const float*)d_in[2];
  const float*   cache    = (const float*)d_in[3];
  const uint8_t* mask     = (const uint8_t*)d_in[4];
  const float*   Wq = (const float*)d_in[5];
  const float*   bq = (const float*)d_in[6];
  const float*   Wk = (const float*)d_in[7];
  const float*   bk = (const float*)d_in[8];
  const float*   Wv = (const float*)d_in[9];
  const float*   bv = (const float*)d_in[10];
  const float*   Wo = (const float*)d_in[11];
  const float*   bo = (const float*)d_in[12];

  // workspace carve-up (bf16 buffers)
  const size_t FF    = (size_t)F_ * F_;                 // 262144
  const size_t NQ    = (size_t)B_ * T1_ * F_;           // 12,320,768
  const size_t NKV   = (size_t)B_ * T2_ * F_;           // 16,908,288
  const size_t NVT   = (size_t)B_ * H_ * DK_ * T2P_;    // 18,874,368

  char* ws = (char*)d_ws;
  size_t off = 0;
  auto alloc = [&](size_t bytes) {
    char* p = ws + off;
    off += (bytes + 255) & ~(size_t)255;
    return p;
  };
  bf16_t* wWq = (bf16_t*)alloc(FF * 2);
  bf16_t* wWk = (bf16_t*)alloc(FF * 2);
  bf16_t* wWv = (bf16_t*)alloc(FF * 2);
  bf16_t* wWo = (bf16_t*)alloc(FF * 2);
  bf16_t* xq  = (bf16_t*)alloc(NQ * 2);
  bf16_t* xk  = (bf16_t*)alloc(NKV * 2);
  bf16_t* xv  = (bf16_t*)alloc(NKV * 2);
  bf16_t* Qb  = (bf16_t*)alloc(NQ * 2);     // (B,H,T1,DK)
  bf16_t* Kb  = (bf16_t*)alloc(NKV * 2);    // (B,H,T2,DK)
  bf16_t* VTb = (bf16_t*)alloc(NVT * 2);    // (B,H,DK,T2P)
  bf16_t* ctx = (bf16_t*)alloc(NQ * 2);     // (B*T1, F)

  const int CT = 256;
  // weight + activation conversion to bf16
  cvt_f32_bf16_kernel<<<(int)((FF  + CT - 1) / CT), CT, 0, stream>>>(Wq, wWq, (int)FF);
  cvt_f32_bf16_kernel<<<(int)((FF  + CT - 1) / CT), CT, 0, stream>>>(Wk, wWk, (int)FF);
  cvt_f32_bf16_kernel<<<(int)((FF  + CT - 1) / CT), CT, 0, stream>>>(Wv, wWv, (int)FF);
  cvt_f32_bf16_kernel<<<(int)((FF  + CT - 1) / CT), CT, 0, stream>>>(Wo, wWo, (int)FF);
  cvt_f32_bf16_kernel<<<(int)((NQ  + CT - 1) / CT), CT, 0, stream>>>(query, xq, (int)NQ);
  pack_kv_kernel     <<<(int)((NKV + CT - 1) / CT), CT, 0, stream>>>(cache, key_in, value_in, xk, xv);

  // zero padded V^T (tail columns t2 in [T2, T2P) must be 0)
  hipMemsetAsync(VTb, 0, NVT * 2, stream);

  // projections (strip counts are exact multiples of 8 waves/block)
  constexpr int stripsQ = (B_ * T1_ / 16) * (F_ / 64);  // 12032
  constexpr int stripsK = (B_ * T2_ / 16) * (F_ / 64);  // 16512
  gemm_wmma_kernel<0><<<stripsQ / 8, 256, 0, stream>>>(xq, wWq, bq, Qb);
  gemm_wmma_kernel<1><<<stripsK / 8, 256, 0, stream>>>(xk, wWk, bk, Kb);
  gemm_wmma_kernel<2><<<stripsK / 8, 256, 0, stream>>>(xv, wWv, bv, VTb);

  // attention
  dim3 agrid(B_ * H_, (T1_ + 15) / 16);                 // (1024, 12)
  attn_kernel<<<agrid, 32, 0, stream>>>(Qb, Kb, VTb, mask, ctx);

  // output projection -> fp32
  gemm_wmma_kernel<3><<<stripsQ / 8, 256, 0, stream>>>(ctx, wWo, bo, d_out);
}
